// Attention_77790447665886
// MI455X (gfx1250) — matrix operand order repaired and checked by reading, not compiled, not run on hardware
//
#include <hip/hip_runtime.h>
#include <hip/hip_bf16.h>

#define B_ 32
#define S_ 2048
#define H_ 512
#define U_ 512

typedef __attribute__((ext_vector_type(16))) __bf16 v16bf;
typedef __attribute__((ext_vector_type(8)))  float  v8f;
typedef __attribute__((ext_vector_type(4)))  float  v4f;

// Branch-free tanh: one v_exp_f32 + one v_rcp_f32, no EXEC divergence.
// tanh(x) = sign(x) * (1 - e) / (1 + e),  e = exp(-2|x|)
__device__ __forceinline__ float fast_tanh(float x) {
    const float ax = __builtin_fabsf(x);
    const float e  = __expf(-2.0f * ax);             // v_exp_f32 path
    const float t  = (1.0f - e) * __builtin_amdgcn_rcpf(1.0f + e);
    return __builtin_copysignf(t, x);
}

// ---------------------------------------------------------------------------
// Prep 1: WwT[u][h] = (bf16) Ww[h][u]   (transpose + convert, LDS-tiled)
// grid: 256 blocks (16x16 tiles of 32x32), block: 256
// ---------------------------------------------------------------------------
__global__ __launch_bounds__(256) void prep_wwt(const float* __restrict__ Ww,
                                                __bf16* __restrict__ WwT) {
    __shared__ float tile[32][33];
    const int tU = blockIdx.x & 15;      // tile col (u)
    const int tH = blockIdx.x >> 4;      // tile row (h)
    const int tx = threadIdx.x & 31;
    const int ty = threadIdx.x >> 5;     // 0..7
    #pragma unroll
    for (int r = 0; r < 32; r += 8) {
        const int h = tH * 32 + ty + r;
        const int u = tU * 32 + tx;
        tile[ty + r][tx] = Ww[h * U_ + u];           // coalesced over u
    }
    __syncthreads();
    #pragma unroll
    for (int r = 0; r < 32; r += 8) {
        const int u = tU * 32 + ty + r;
        const int h = tH * 32 + tx;
        WwT[(size_t)u * H_ + h] = (__bf16)tile[tx][ty + r];  // coalesced over h
    }
}

// ---------------------------------------------------------------------------
// Prep 2: comb[b][u] = Wb[u] + W1b[u] + sum_h dec[b][h] * W1w[h][u]
// grid: 32 (one per batch), block: 256 (each thread covers u and u+256)
// ---------------------------------------------------------------------------
__global__ __launch_bounds__(256) void prep_comb(const float* __restrict__ dec,
                                                 const float* __restrict__ W1w,
                                                 const float* __restrict__ Wb,
                                                 const float* __restrict__ W1b,
                                                 float* __restrict__ comb) {
    const int b = blockIdx.x;
    const int u = threadIdx.x;
    float a0 = 0.f, a1 = 0.f;
    for (int h = 0; h < H_; ++h) {
        const float d = dec[b * H_ + h];
        a0 += d * W1w[h * U_ + u];
        a1 += d * W1w[h * U_ + u + 256];
    }
    comb[b * U_ + u]       = a0 + Wb[u]       + W1b[u];
    comb[b * U_ + u + 256] = a1 + Wb[u + 256] + W1b[u + 256];
}

// ---------------------------------------------------------------------------
// Main fused kernel: 64-row slab per WG (4 M-tiles of 16):
//   enc_proj = enc_slab @ Ww   (bf16 WMMA, fp32 accum, K=512, N=512)
//   unnorm   = tanh(enc_proj + comb[b]) . vw + vb
// grid: B*S/64 = 1024, block: 256 (8 waves); wave w owns columns [64w,64w+64)
// Each wave: 4x4 grid of 16x16 f32 accumulators; every B-fragment load is
// reused by 4 WMMAs (amortizes Ww L2 traffic 4x vs a 16-row slab).
// ---------------------------------------------------------------------------
__global__ __launch_bounds__(256) void attn_gemm_kernel(
    const float*  __restrict__ enc,    // [B,S,H]
    const __bf16* __restrict__ WwT,    // [U,H]  (bf16, K-contiguous per column)
    const float*  __restrict__ comb,   // [B,U]
    const float*  __restrict__ vw,     // [U]
    const float*  __restrict__ vb,     // [1]
    float* __restrict__ unnorm_out)    // [B,S]
{
    __shared__ float red[8][64];       // per-wave row partials (deterministic)

    const int tid  = threadIdx.x;
    const int wave = tid >> 5;
    const int lane = tid & 31;
    const int nl   = lane & 15;        // 0..15
    const int hh   = lane >> 4;        // 0 or 1 (K-half)

    const int wg = blockIdx.x;
    const int b  = wg >> 5;                 // / (S/64)
    const int s0 = (wg & 31) << 6;          // * 64

    const int n_base = wave * 64;

    // A row pointers for the 4 M-tiles (lane nl -> row m = nl within tile)
    const float* aptr[4];
    #pragma unroll
    for (int mt = 0; mt < 4; ++mt)
        aptr[mt] = enc + ((size_t)(b * S_ + s0 + mt * 16 + nl)) * H_;

    v8f acc[4][4] = {};   // [mtile][ntile]

    for (int k0 = 0; k0 < H_; k0 += 32) {
        // ---- A fragments (16x32; this code's contiguous K order, same as B):
        // lane = m + 16*hh holds K = k0+16*hh+[0..15] (contiguous)
        v16bf afrag[4];
        #pragma unroll
        for (int mt = 0; mt < 4; ++mt) {
            const v4f a0 = *(const v4f*)(aptr[mt] + k0 + hh * 16);
            const v4f a1 = *(const v4f*)(aptr[mt] + k0 + hh * 16 + 4);
            const v4f a2 = *(const v4f*)(aptr[mt] + k0 + hh * 16 + 8);
            const v4f a3 = *(const v4f*)(aptr[mt] + k0 + hh * 16 + 12);
            #pragma unroll
            for (int i = 0; i < 4; ++i) {
                afrag[mt][i]      = (__bf16)a0[i];
                afrag[mt][4 + i]  = (__bf16)a1[i];
                afrag[mt][8 + i]  = (__bf16)a2[i];
                afrag[mt][12 + i] = (__bf16)a3[i];
            }
        }
        // ---- B fragments (32x16; this code's contiguous K order, same as A):
        // lane = n + 16*hh holds K = k0+16*hh+[0..15] of column n (contiguous)
        #pragma unroll
        for (int nt = 0; nt < 4; ++nt) {
            const __bf16* bp = WwT + (size_t)(n_base + nt * 16 + nl) * H_ + k0 + hh * 16;
            const v16bf bfrag = *(const v16bf*)bp;
            #pragma unroll
            for (int mt = 0; mt < 4; ++mt) {
                acc[mt][nt] = __builtin_amdgcn_wmma_f32_16x16x32_bf16(
                    false, afrag[mt], false, bfrag, (short)0, acc[mt][nt], false, false);
            }
        }
    }

    // ---- Epilogue: tanh(acc + comb) * vw, reduce over this wave's 64 columns.
    // C/D layout: VGPR r of lane l -> (m = r + 8*(l>>4), n = n_base+16t+(l&15))
    #pragma unroll
    for (int mt = 0; mt < 4; ++mt) {
        float rowp[8];
        #pragma unroll
        for (int r = 0; r < 8; ++r) rowp[r] = 0.f;
        #pragma unroll
        for (int nt = 0; nt < 4; ++nt) {
            const int n   = n_base + nt * 16 + nl;
            const float c = comb[b * U_ + n];
            const float v = vw[n];
            #pragma unroll
            for (int r = 0; r < 8; ++r)
                rowp[r] += fast_tanh(acc[mt][nt][r] + c) * v;
        }
        // Butterfly sum across the 16 lanes of each half (masks keep halves apart)
        #pragma unroll
        for (int m = 1; m < 16; m <<= 1) {
            #pragma unroll
            for (int r = 0; r < 8; ++r)
                rowp[r] += __shfl_xor(rowp[r], m, 32);
        }
        if (nl == 0) {
            #pragma unroll
            for (int r = 0; r < 8; ++r)
                red[wave][mt * 16 + hh * 8 + r] = rowp[r];
        }
    }
    __syncthreads();
    if (tid < 64) {
        float s = 0.f;
        #pragma unroll
        for (int w = 0; w < 8; ++w) s += red[w][tid];   // fixed order: deterministic
        unnorm_out[(size_t)b * S_ + s0 + tid] = s + vb[0];
    }
}

// ---------------------------------------------------------------------------
// Softmax over S per batch.  grid: 32, block: 256 (8 values per thread)
// ---------------------------------------------------------------------------
__global__ __launch_bounds__(256) void softmax_kernel(const float* __restrict__ unnorm,
                                                      float* __restrict__ attn) {
    __shared__ float sred[256];
    const int b = blockIdx.x, tid = threadIdx.x;
    const float* u = unnorm + (size_t)b * S_;
    float vals[8];
    float mx = -INFINITY;
    #pragma unroll
    for (int i = 0; i < 8; ++i) { vals[i] = u[tid + i * 256]; mx = fmaxf(mx, vals[i]); }
    sred[tid] = mx;
    __syncthreads();
    for (int off = 128; off > 0; off >>= 1) {
        if (tid < off) sred[tid] = fmaxf(sred[tid], sred[tid + off]);
        __syncthreads();
    }
    mx = sred[0];
    __syncthreads();
    float sum = 0.f;
    #pragma unroll
    for (int i = 0; i < 8; ++i) { vals[i] = __expf(vals[i] - mx); sum += vals[i]; }
    sred[tid] = sum;
    __syncthreads();
    for (int off = 128; off > 0; off >>= 1) {
        if (tid < off) sred[tid] += sred[tid + off];
        __syncthreads();
    }
    const float inv = 1.0f / sred[0];
    #pragma unroll
    for (int i = 0; i < 8; ++i)
        attn[(size_t)b * S_ + tid + i * 256] = vals[i] * inv;
}

// ---------------------------------------------------------------------------
// c_vec partials: WG (b, chunk c of 256 s-values):
//   part[c][b][h] = sum_{s in chunk} attn[b][s] * enc[b][s][h]
// grid: 256, block: 256 (thread covers h and h+256, coalesced over h)
// ---------------------------------------------------------------------------
__global__ __launch_bounds__(256) void cvec_partial(const float* __restrict__ attn,
                                                    const float* __restrict__ enc,
                                                    float* __restrict__ part) {
    __shared__ float sa[256];
    const int b = blockIdx.x >> 3;
    const int c = blockIdx.x & 7;
    const int tid = threadIdx.x;
    const int sbase = c * 256;
    sa[tid] = attn[(size_t)b * S_ + sbase + tid];
    __syncthreads();
    const float* e = enc + ((size_t)b * S_ + sbase) * H_;
    float a0 = 0.f, a1 = 0.f;
    for (int s = 0; s < 256; ++s) {
        const float a = sa[s];
        a0 += a * e[(size_t)s * H_ + tid];
        a1 += a * e[(size_t)s * H_ + tid + 256];
    }
    part[((size_t)c * B_ + b) * H_ + tid]       = a0;
    part[((size_t)c * B_ + b) * H_ + tid + 256] = a1;
}

// grid: 64, block: 256  -> c_vec[b][h] = sum_c part[c][b][h]  (fixed order)
__global__ __launch_bounds__(256) void cvec_reduce(const float* __restrict__ part,
                                                   float* __restrict__ cvec) {
    const int i = blockIdx.x * 256 + threadIdx.x;   // [0, B*H)
    const int b = i >> 9, h = i & 511;
    float s = 0.f;
    #pragma unroll
    for (int c = 0; c < 8; ++c) s += part[((size_t)c * B_ + b) * H_ + h];
    cvec[i] = s;
}

// ---------------------------------------------------------------------------
extern "C" void kernel_launch(void* const* d_in, const int* in_sizes, int n_in,
                              void* d_out, int out_size, void* d_ws, size_t ws_size,
                              hipStream_t stream) {
    const float* enc = (const float*)d_in[0];   // [32,2048,512]
    const float* dec = (const float*)d_in[1];   // [32,512]
    const float* Ww  = (const float*)d_in[2];   // [512,512]
    const float* Wb  = (const float*)d_in[3];   // [512]
    const float* W1w = (const float*)d_in[4];   // [512,512]
    const float* W1b = (const float*)d_in[5];   // [512]
    const float* vw  = (const float*)d_in[6];   // [512,1]
    const float* vb  = (const float*)d_in[7];   // [1]

    float* out    = (float*)d_out;
    float* cvec   = out;                 // [32,512]   = 16384
    float* unnorm = out + 16384;         // [32,2048]  = 65536
    float* attn   = out + 81920;         // [32,2048]  = 65536

    char* ws = (char*)d_ws;
    __bf16* WwT = (__bf16*)ws;                       // 512*512*2 = 512 KiB
    float*  comb = (float*)(ws + 524288);            // 32*512*4  =  64 KiB
    float*  part = (float*)(ws + 589824);            // 8*32*512*4= 512 KiB

    prep_wwt        <<<256,  256, 0, stream>>>(Ww, WwT);
    prep_comb       <<<32,   256, 0, stream>>>(dec, W1w, Wb, W1b, comb);
    attn_gemm_kernel<<<1024, 256, 0, stream>>>(enc, WwT, comb, vw, vb, unnorm);
    softmax_kernel  <<<32,   256, 0, stream>>>(unnorm, attn);
    cvec_partial    <<<256,  256, 0, stream>>>(attn, enc, part);
    cvec_reduce     <<<64,   256, 0, stream>>>(part, cvec);
}
